// Layer_GCN_58471684768393
// MI455X (gfx1250) — compile-verified, matrix operand order checked by reading, not run on hardware
//
#include <hip/hip_runtime.h>
#include <hip/hip_bf16.h>
#include <hip/hip_runtime_api.h>

// ---------------------------------------------------------------------------
// GCN layer for MI455X (gfx1250):
//   support = X @ W   -> bf16 WMMA, W staged through LDS with async-to-LDS DMA
//   out = segsum(support[src]*w, dst) + bias -> per-wave scatter, f32 atomics
// ---------------------------------------------------------------------------

#define D_DIM 256   // D_IN == D_OUT == 256

typedef __attribute__((ext_vector_type(16))) __bf16 v16bf;
typedef __attribute__((ext_vector_type(8)))  float  v8f;

union Frag16 {
    v16bf v;
    uint4 q[2];
    unsigned u[8];
};

__device__ __forceinline__ unsigned pack2bf(float a, float b) {
    // round-to-nearest-even f32 -> bf16, packed low/high
    unsigned ua = __float_as_uint(a);
    unsigned ub = __float_as_uint(b);
    unsigned ra = (ua + 0x7FFFu + ((ua >> 16) & 1u)) >> 16;
    unsigned rb = (ub + 0x7FFFu + ((ub >> 16) & 1u)) >> 16;
    return (ra & 0xFFFFu) | (rb << 16);
}

__device__ __forceinline__ void async_load_b128_to_lds(unsigned lds_byte_addr,
                                                       const void* gaddr) {
    unsigned long long ga = (unsigned long long)(size_t)gaddr;
    // VDST = per-lane LDS byte address, VADDR = 64-bit global address.
    asm volatile("global_load_async_to_lds_b128 %0, %1, off"
                 :: "v"(lds_byte_addr), "v"(ga) : "memory");
}

__device__ __forceinline__ void wait_asynccnt0() {
#if __has_builtin(__builtin_amdgcn_s_wait_asynccnt)
    __builtin_amdgcn_s_wait_asynccnt(0);
#else
    asm volatile("s_wait_asynccnt 0x0" ::: "memory");
#endif
}

// ---------------------------------------------------------------------------
// Kernel 0: convert W (256x256 f32 row-major, W[k][n]) into bf16 B-fragment
// layout: [kb(8)][t(16)][lane(32)][dword(8)]  -- each kb slice is a contiguous
// 16 KB block so the GEMM can async-DMA one K-slice at a time into LDS.
// B fragment (column-major 32x16 bf16, mirrors the A 16x32 layout):
//   lane<16 : col n = n0+lane,   dwords 0-3 -> k0+{0..7},   4-7 -> k0+16+{0..7}
//   lane>=16: col n = n0+lane-16, dwords 0-3 -> k0+8+{0..7}, 4-7 -> k0+24+{0..7}
// ---------------------------------------------------------------------------
__global__ __launch_bounds__(256) void convw_kernel(const float* __restrict__ W,
                                                    unsigned* __restrict__ wswz) {
    int tid = blockIdx.x * blockDim.x + threadIdx.x;     // 0 .. 32767
    if (tid >= 8 * 16 * 32 * 8) return;
    int d    = tid & 7;
    int lane = (tid >> 3) & 31;
    int t    = (tid >> 8) & 15;
    int kb   = tid >> 12;

    int n    = t * 16 + (lane & 15);
    int half = lane >> 4;
    int seg  = d >> 2;
    int k    = kb * 32 + half * 8 + seg * 16 + (d & 3) * 2;

    wswz[tid] = pack2bf(W[(size_t)k * D_DIM + n], W[(size_t)(k + 1) * D_DIM + n]);
}

// ---------------------------------------------------------------------------
// Kernel 1: support = X @ W via bf16 WMMA.
// Block = 128 threads (4 waves). Each wave computes a 16-row x 256-col strip.
// W K-slices (16 KB each) are double-buffered in LDS via async-to-LDS DMA:
// slice kb+1 streams in while slice kb feeds the WMMAs through ds_load_b128.
// X streams from HBM exactly once.
// ---------------------------------------------------------------------------
__global__ __launch_bounds__(128) void gemm_kernel(const float* __restrict__ x,
                                                   const unsigned* __restrict__ wswz,
                                                   float* __restrict__ support,
                                                   int nNodes) {
    __shared__ unsigned wlds[2][4096];   // 2 x 16 KB K-slices

    const int tid  = threadIdx.x;
    const int lane = tid & 31;
    const int wv   = tid >> 5;
    const int row0 = blockIdx.x * 64 + wv * 16;
    const int half = lane >> 4;
    const int l15  = lane & 15;

    int rowA = row0 + l15;
    if (rowA >= nNodes) rowA = nNodes - 1;   // clamp: WMMA needs EXEC all-1s

    // ---- stage slice kb into LDS buffer buf (1024 x b128, 8 per thread) ----
    auto stage = [&](int kb, int buf) {
        const uint4* gsrc = (const uint4*)(wswz + (size_t)kb * 4096);
        unsigned lbase = (unsigned)(size_t)&wlds[buf][0];
#pragma unroll
        for (int j = 0; j < 8; ++j) {
            int idx = j * 128 + tid;
            async_load_b128_to_lds(lbase + idx * 16, gsrc + idx);
        }
    };

    v8f acc[16];
    v8f zero = {0.f, 0.f, 0.f, 0.f, 0.f, 0.f, 0.f, 0.f};
#pragma unroll
    for (int t = 0; t < 16; ++t) acc[t] = zero;

    stage(0, 0);
    wait_asynccnt0();
    __syncthreads();

    for (int kb = 0; kb < 8; ++kb) {
        const int cur = kb & 1;
        if (kb < 7) stage(kb + 1, cur ^ 1);   // async: overlaps with compute

        // ---- A fragment: 16x32 bf16 (this wave's rows, K = kb*32..+31) ----
        Frag16 a;
        const float* ap = x + (size_t)rowA * D_DIM + kb * 32 + half * 8;
        float4 p0 = *(const float4*)(ap);
        float4 p1 = *(const float4*)(ap + 4);
        float4 p2 = *(const float4*)(ap + 16);
        float4 p3 = *(const float4*)(ap + 20);
        a.u[0] = pack2bf(p0.x, p0.y); a.u[1] = pack2bf(p0.z, p0.w);
        a.u[2] = pack2bf(p1.x, p1.y); a.u[3] = pack2bf(p1.z, p1.w);
        a.u[4] = pack2bf(p2.x, p2.y); a.u[5] = pack2bf(p2.z, p2.w);
        a.u[6] = pack2bf(p3.x, p3.y); a.u[7] = pack2bf(p3.z, p3.w);

        // ---- 16 column tiles, B double-buffered from LDS ----
        Frag16 bf[2];
        {
            const uint4* p = (const uint4*)&wlds[cur][lane * 8];
            bf[0].q[0] = p[0];
            bf[0].q[1] = p[1];
        }
#pragma unroll
        for (int t = 0; t < 16; ++t) {
            if (t < 15) {
                const uint4* p = (const uint4*)&wlds[cur][((t + 1) * 32 + lane) * 8];
                bf[(t + 1) & 1].q[0] = p[0];
                bf[(t + 1) & 1].q[1] = p[1];
            }
            acc[t] = __builtin_amdgcn_wmma_f32_16x16x32_bf16(
                false, a.v, false, bf[t & 1].v, (short)0, acc[t], false, false);
        }

        if (kb < 7) {
            wait_asynccnt0();     // staging of next slice done (this wave)
            __syncthreads();      // all waves done reading + staging
        }
    }

    // ---- store C: lane<16 -> rows row0+0..7, lane>=16 -> rows row0+8..15 ----
    const int r0 = row0 + half * 8;
#pragma unroll
    for (int t = 0; t < 16; ++t) {
        int col = t * 16 + l15;
#pragma unroll
        for (int j = 0; j < 8; ++j) {
            int row = r0 + j;
            if (row < nNodes) support[(size_t)row * D_DIM + col] = acc[t][j];
        }
    }
}

// ---------------------------------------------------------------------------
// Kernel 2: out[i][d] = bias[d]  (float4 stores; scatter atomics accumulate)
// ---------------------------------------------------------------------------
__global__ __launch_bounds__(256) void init_out_kernel(float* __restrict__ out,
                                                       const float* __restrict__ bias,
                                                       int total4) {
    int idx = blockIdx.x * blockDim.x + threadIdx.x;
    if (idx >= total4) return;
    int col4 = (idx & (D_DIM / 4 - 1)) * 4;
    ((float4*)out)[idx] = *(const float4*)(bias + col4);
}

// ---------------------------------------------------------------------------
// Kernel 3: per-edge scatter. One wave32 per edge: lanes cover the 256-wide
// row (8 floats each), two float4 gathers from L2-resident support, 8 native
// f32 atomic adds. Prefetch the next edge's source row into cache.
// ---------------------------------------------------------------------------
__global__ __launch_bounds__(256) void scatter_kernel(const float* __restrict__ support,
                                                      const int* __restrict__ esrc,
                                                      const int* __restrict__ edst,
                                                      const float* __restrict__ ew,
                                                      float* __restrict__ out,
                                                      int nEdges) {
    const int lane  = threadIdx.x & 31;
    const int wavesPerBlock = blockDim.x >> 5;
    const int gw = __builtin_amdgcn_readfirstlane(
        (int)(blockIdx.x * wavesPerBlock + (threadIdx.x >> 5)));
    const int totalWaves = gridDim.x * wavesPerBlock;

    for (int e = gw; e < nEdges; e += totalWaves) {
        int   s  = __builtin_amdgcn_readfirstlane(esrc[e]);
        int   d  = __builtin_amdgcn_readfirstlane(edst[e]);
        float w  = __uint_as_float(__builtin_amdgcn_readfirstlane(__float_as_uint(ew[e])));

        // prefetch next edge's source row (gfx1250 global_prefetch_b8)
        int en = e + totalWaves;
        if (en < nEdges) {
            int sn = esrc[en];
            __builtin_prefetch(support + (size_t)sn * D_DIM + lane * 8, 0, 0);
        }

        const float* sp = support + (size_t)s * D_DIM + lane * 8;
        float4 v0 = *(const float4*)(sp);
        float4 v1 = *(const float4*)(sp + 4);

        float* op = out + (size_t)d * D_DIM + lane * 8;
        unsafeAtomicAdd(op + 0, v0.x * w);
        unsafeAtomicAdd(op + 1, v0.y * w);
        unsafeAtomicAdd(op + 2, v0.z * w);
        unsafeAtomicAdd(op + 3, v0.w * w);
        unsafeAtomicAdd(op + 4, v1.x * w);
        unsafeAtomicAdd(op + 5, v1.y * w);
        unsafeAtomicAdd(op + 6, v1.z * w);
        unsafeAtomicAdd(op + 7, v1.w * w);
    }
}

// ---------------------------------------------------------------------------
extern "C" void kernel_launch(void* const* d_in, const int* in_sizes, int n_in,
                              void* d_out, int out_size, void* d_ws, size_t ws_size,
                              hipStream_t stream) {
    const float* x    = (const float*)d_in[0];
    const int*   esrc = (const int*)d_in[1];
    const int*   edst = (const int*)d_in[2];
    const float* ew   = (const float*)d_in[3];
    const float* W    = (const float*)d_in[4];
    const float* bias = (const float*)d_in[5];
    float*       out  = (float*)d_out;

    const int nNodes = in_sizes[0] / D_DIM;
    const int nEdges = in_sizes[1];

    // workspace: [ support f32 (nNodes*256) | swizzled bf16 W (32768 dwords) ]
    float*    support = (float*)d_ws;
    unsigned* wswz    = (unsigned*)((char*)d_ws + (size_t)nNodes * D_DIM * sizeof(float));

    // 0) W -> bf16 B-fragment layout, contiguous per-kb slices (32768 dwords)
    convw_kernel<<<128, 256, 0, stream>>>(W, wswz);

    // 1) support = X @ W  (WMMA, async-to-LDS double-buffered W slices)
    gemm_kernel<<<(nNodes + 63) / 64, 128, 0, stream>>>(x, wswz, support, nNodes);

    // 2) out = bias (broadcast)
    int total4 = nNodes * (D_DIM / 4);
    init_out_kernel<<<(total4 + 255) / 256, 256, 0, stream>>>(out, bias, total4);

    // 3) scatter-add over edges
    scatter_kernel<<<8192, 256, 0, stream>>>(support, esrc, edst, ew, out, nEdges);
}
